// ResidualGNN1_47708496724555
// MI455X (gfx1250) — compile-verified, matrix operand order checked by reading, not run on hardware
//
#include <hip/hip_runtime.h>
#include <hip/hip_bf16.h>

#define NNODES 50000
#define NEDGES 800000
#define EPSBN  1e-5f
#define SLOPE  0.01f

typedef __bf16 bf16_t;
typedef __attribute__((ext_vector_type(16))) __bf16 v16bf;
typedef __attribute__((ext_vector_type(8)))  __bf16 v8bf;
typedef __attribute__((ext_vector_type(8)))  float  v8f;

__device__ inline bf16_t f2bf(float f) {
    unsigned u = __builtin_bit_cast(unsigned, f);
    unsigned r = u + 0x7FFFu + ((u >> 16) & 1u);   // round-to-nearest-even
    unsigned short h = (unsigned short)(r >> 16);
    return __builtin_bit_cast(bf16_t, h);
}
__device__ inline float bf2f(bf16_t b) {
    unsigned short u = __builtin_bit_cast(unsigned short, b);
    return __builtin_bit_cast(float, ((unsigned)u) << 16);
}

// ---------------- degree / norm ----------------
__global__ void k_deg_init(float* deg) {
    int i = blockIdx.x * blockDim.x + threadIdx.x;
    if (i < NNODES) deg[i] = 1.0f;                 // self-loop
}
__global__ void k_deg_count(const int* __restrict__ ei, float* deg) {
    int e = blockIdx.x * blockDim.x + threadIdx.x;
    if (e >= NEDGES) return;
    int d = ei[NEDGES + e];
    d = d < 0 ? 0 : (d >= NNODES ? NNODES - 1 : d);
    atomicAdd(&deg[d], 1.0f);
}
__global__ void k_deg_rsqrt(float* deg) {
    int i = blockIdx.x * blockDim.x + threadIdx.x;
    if (i < NNODES) deg[i] = rsqrtf(deg[i]);
}

// ---------------- f32 -> bf16 convert ----------------
__global__ void k_f2bf(const float* __restrict__ src, bf16_t* __restrict__ dst, long long n) {
    long long i = (long long)blockIdx.x * blockDim.x + threadIdx.x;
    if (i < n) dst[i] = f2bf(src[i]);
}

// ---------------- W repack into WMMA B-fragment order ----------------
// Fragment index layout per lane: elements 0..7 -> K = kt*32 + off + {0..7},
// elements 8..15 -> K = kt*32 + 16 + off + {0..7}, off = (lane<16)?0:8, col = ct*16 + (lane&15)
__global__ void k_repack_w(const float* __restrict__ W, bf16_t* __restrict__ Wp, int K, int dout) {
    int KT = K >> 5, CT = dout >> 4;
    int idx = blockIdx.x * blockDim.x + threadIdx.x;
    int total = KT * CT * 32;
    if (idx >= total) return;
    int lane = idx & 31;
    int t    = idx >> 5;
    int kt   = t % KT;
    int ct   = t / KT;
    int col  = ct * 16 + (lane & 15);
    int off  = (lane < 16) ? 0 : 8;
    union { bf16_t s[16]; v16bf v; } u;
#pragma unroll
    for (int i = 0; i < 16; ++i) {
        int k = kt * 32 + ((i < 8) ? (i + off) : (i + 8 + off));
        u.s[i] = f2bf(W[(size_t)k * dout + col]);
    }
    v16bf* outp = reinterpret_cast<v16bf*>(Wp);
    outp[(size_t)(ct * KT + kt) * 32 + lane] = u.v;
}

// ---------------- WMMA GEMM: C[N,dout] = A[N,K](bf16) @ W(bf16 packed) ----------------
// One wave -> 16 rows x 64 cols (4 wmma tiles, A fragment reused 4x per K-step).
__global__ __launch_bounds__(256) void k_wmma_gemm(const bf16_t* __restrict__ A,
                                                   const bf16_t* __restrict__ Wp,
                                                   float* __restrict__ C,
                                                   int K, int dout) {
    const int lane = threadIdx.x & 31;
    const int wave = threadIdx.x >> 5;
    const int m0   = (blockIdx.x * 8 + wave) * 16;
    const int CT   = dout >> 4;
    const int KT   = K >> 5;
    const int ct0  = blockIdx.y * 4;

    int mrow = m0 + (lane & 15);
    if (mrow >= NNODES) mrow = NNODES - 1;
    const int off = (lane < 16) ? 0 : 8;

    const v8bf*  rowp = reinterpret_cast<const v8bf*>(A + (size_t)mrow * K);
    const v16bf* wp   = reinterpret_cast<const v16bf*>(Wp);

    v8f acc[4] = {};
    int ctc[4];
#pragma unroll
    for (int t = 0; t < 4; ++t) { int c = ct0 + t; ctc[t] = (c < CT) ? c : (CT - 1); }

    for (int kt = 0; kt < KT; ++kt) {
        const int kb = kt << 5;
        union { v16bf v; v8bf h[2]; } af;
        af.h[0] = rowp[(kb + off) >> 3];
        af.h[1] = rowp[(kb + 16 + off) >> 3];
#pragma unroll
        for (int t = 0; t < 4; ++t) {
            v16bf bfrag = wp[((size_t)ctc[t] * KT + kt) * 32 + lane];
            acc[t] = __builtin_amdgcn_wmma_f32_16x16x32_bf16(
                false, af.v, false, bfrag, (short)0, acc[t], false, false);
        }
    }

    const int colb  = lane & 15;
    const int rbase = (lane < 16) ? 0 : 8;
#pragma unroll
    for (int t = 0; t < 4; ++t) {
        if (ct0 + t >= CT) break;
        int col = (ct0 + t) * 16 + colb;
#pragma unroll
        for (int r = 0; r < 8; ++r) {
            int row = m0 + rbase + r;
            if (row < NNODES) C[(size_t)row * dout + col] = acc[t][r];
        }
    }
}

// ---------------- agg init: self-loop term + bias ----------------
__global__ void k_agg_init(const float* __restrict__ hw, const float* __restrict__ dinv,
                           const float* __restrict__ bias, float* __restrict__ agg, int dout) {
    long long idx = (long long)blockIdx.x * blockDim.x + threadIdx.x;
    long long total = (long long)NNODES * dout;
    if (idx >= total) return;
    int col = (int)(idx & (long long)(dout - 1));       // dout is a power of two
    int row = (int)(idx >> __builtin_ctz(dout));
    float di = dinv[row];
    agg[idx] = hw[idx] * di * di + bias[col];
}

// ---------------- edge scatter with f32 atomics + next-iteration prefetch ----------------
__global__ __launch_bounds__(256) void k_scatter(const int* __restrict__ ei,
                                                 const float* __restrict__ dinv,
                                                 const float* __restrict__ hw,
                                                 float* __restrict__ agg,
                                                 int dout, int cshift) {
    const int chunks = dout >> 4;                        // 16 floats per chunk
    const long long total  = (long long)NEDGES * chunks;
    const long long stride = (long long)gridDim.x * blockDim.x;
    for (long long idx = (long long)blockIdx.x * blockDim.x + threadIdx.x;
         idx < total; idx += stride) {
        // warm L2 with the next iteration's gather row while this one's atomics drain
        long long nidx = idx + stride;
        if (nidx < total) {
            int en = (int)(nidx >> cshift);
            int cn = (int)(nidx & (long long)(chunks - 1));
            int sn = ei[en];
            sn = sn < 0 ? 0 : (sn >= NNODES ? NNODES - 1 : sn);
            __builtin_prefetch(hw + (size_t)sn * dout + cn * 16, 0, 3);
        }
        int e = (int)(idx >> cshift);
        int c = (int)(idx & (long long)(chunks - 1));
        int s = ei[e];
        int d = ei[NEDGES + e];
        s = s < 0 ? 0 : (s >= NNODES ? NNODES - 1 : s);
        d = d < 0 ? 0 : (d >= NNODES ? NNODES - 1 : d);
        float nrm = dinv[s] * dinv[d];
        const float4* src = reinterpret_cast<const float4*>(hw + (size_t)s * dout + c * 16);
        float* dstp = agg + (size_t)d * dout + c * 16;
#pragma unroll
        for (int q = 0; q < 4; ++q) {
            float4 v = src[q];
            atomicAdd(dstp + q * 4 + 0, v.x * nrm);
            atomicAdd(dstp + q * 4 + 1, v.y * nrm);
            atomicAdd(dstp + q * 4 + 2, v.z * nrm);
            atomicAdd(dstp + q * 4 + 3, v.w * nrm);
        }
    }
}

// ---------------- BN: zero / partial (LDS atomics) / final ----------------
__global__ void k_zero2(float* a, float* b, int n) {
    int i = blockIdx.x * blockDim.x + threadIdx.x;
    if (i < n) { a[i] = 0.f; b[i] = 0.f; }
}
__global__ __launch_bounds__(256) void k_bn_partial(const float* __restrict__ agg,
                                                    float* __restrict__ sum,
                                                    float* __restrict__ sumsq, int dout) {
    __shared__ float ls[512];
    __shared__ float lq[512];
    for (int j = threadIdx.x; j < dout; j += blockDim.x) { ls[j] = 0.f; lq[j] = 0.f; }
    __syncthreads();
    long long total  = (long long)NNODES * dout;
    long long stride = (long long)gridDim.x * blockDim.x;
    for (long long idx = (long long)blockIdx.x * blockDim.x + threadIdx.x; idx < total; idx += stride) {
        float v = agg[idx];
        int col = (int)(idx & (long long)(dout - 1));
        atomicAdd(&ls[col], v);
        atomicAdd(&lq[col], v * v);
    }
    __syncthreads();
    for (int j = threadIdx.x; j < dout; j += blockDim.x) {
        atomicAdd(&sum[j], ls[j]);
        atomicAdd(&sumsq[j], lq[j]);
    }
}
__global__ void k_bn_final(float* sum, float* sumsq, int dout) {
    int j = blockIdx.x * blockDim.x + threadIdx.x;
    if (j >= dout) return;
    float mu  = sum[j] / (float)NNODES;
    float var = sumsq[j] / (float)NNODES - mu * mu;     // biased var, matches BN training math
    sum[j]   = mu;
    sumsq[j] = rsqrtf(var + EPSBN);
}

// ---------------- normalize + affine (+residual) + LeakyReLU ----------------
__global__ void k_bn_norm(const float* __restrict__ agg, const float* __restrict__ mu,
                          const float* __restrict__ inv, const float* __restrict__ gamma,
                          const float* __restrict__ beta, const bf16_t* __restrict__ resid,
                          bf16_t* __restrict__ out_bf, bf16_t* __restrict__ out_bf2,
                          float* __restrict__ out_f32, int dout) {
    long long idx = (long long)blockIdx.x * blockDim.x + threadIdx.x;
    long long total = (long long)NNODES * dout;
    if (idx >= total) return;
    int col = (int)(idx & (long long)(dout - 1));
    float v = (agg[idx] - mu[col]) * inv[col] * gamma[col] + beta[col];
    if (resid) v += bf2f(resid[idx]);
    v = v > 0.f ? v : SLOPE * v;
    if (out_bf)  out_bf[idx]  = f2bf(v);
    if (out_bf2) out_bf2[idx] = f2bf(v);
    if (out_f32) out_f32[idx] = v;
}

// ---------------- global MLP branch ----------------
__global__ void k_global(const float* __restrict__ gf, const float* __restrict__ gW1,
                         const float* __restrict__ gb1, const float* __restrict__ gW2,
                         const float* __restrict__ gb2, float* __restrict__ g2) {
    int n = blockIdx.x * blockDim.x + threadIdx.x;
    if (n >= NNODES) return;
    float x = gf[n];
    float acc = gb2[0];
#pragma unroll 8
    for (int h = 0; h < 64; ++h) {
        float t = x * gW1[h] + gb1[h];
        t = t > 0.f ? t : SLOPE * t;
        acc += t * gW2[h];
    }
    g2[n] = acc;
}

// ---------------- final head: concat @ fcW + sigmoid ----------------
__global__ void k_final(const float* __restrict__ act32, const float* __restrict__ g2,
                        const float* __restrict__ fcW, const float* __restrict__ fcb,
                        float* __restrict__ out) {
    int n = blockIdx.x * blockDim.x + threadIdx.x;
    if (n >= NNODES) return;
    float s = fcb[0];
    const float* row = act32 + (size_t)n * 32;
#pragma unroll
    for (int j = 0; j < 32; ++j) s += row[j] * fcW[j];
    s += g2[n] * fcW[32];
    out[n] = 1.0f / (1.0f + __expf(-s));
}

extern "C" void kernel_launch(void* const* d_in, const int* in_sizes, int n_in,
                              void* d_out, int out_size, void* d_ws, size_t ws_size,
                              hipStream_t stream) {
    const float* x  = (const float*)d_in[0];
    const int*   ei = (const int*)d_in[1];
    const float* gf = (const float*)d_in[2];
    const float *W[6], *Bv[6], *Gv[6], *Bt[6];
    for (int i = 0; i < 6; ++i) {
        W[i]  = (const float*)d_in[3 + 4 * i];
        Bv[i] = (const float*)d_in[4 + 4 * i];
        Gv[i] = (const float*)d_in[5 + 4 * i];
        Bt[i] = (const float*)d_in[6 + 4 * i];
    }
    const float* gW1 = (const float*)d_in[27];
    const float* gb1 = (const float*)d_in[28];
    const float* gW2 = (const float*)d_in[29];
    const float* gb2 = (const float*)d_in[30];
    const float* fcW = (const float*)d_in[31];
    const float* fcb = (const float*)d_in[32];
    float* out = (float*)d_out;

    // ---- workspace carve-out (256B aligned) ----
    char* ws = (char*)d_ws;
    size_t off = 0;
    auto carve = [&](size_t bytes) -> void* {
        void* p = ws + off;
        off = (off + bytes + 255) & ~(size_t)255;
        return p;
    };
    float*  dinv    = (float*)carve((size_t)NNODES * 4);
    bf16_t* wpack   = (bf16_t*)carve((size_t)512 * 512 * 2);
    bf16_t* actbf   = (bf16_t*)carve((size_t)NNODES * 512 * 2);
    bf16_t* identbf = (bf16_t*)carve((size_t)NNODES * 512 * 2);
    float*  hw      = (float*)carve((size_t)NNODES * 512 * 4);
    float*  agg     = (float*)carve((size_t)NNODES * 512 * 4);
    float*  bsum    = (float*)carve(512 * 4);
    float*  bsq     = (float*)carve(512 * 4);
    float*  g2      = (float*)carve((size_t)NNODES * 4);
    (void)ws_size; (void)in_sizes; (void)n_in; (void)out_size;

    const int TB = 256;
    // degree / D^-1/2
    k_deg_init<<<(NNODES + TB - 1) / TB, TB, 0, stream>>>(dinv);
    k_deg_count<<<(NEDGES + TB - 1) / TB, TB, 0, stream>>>(ei, dinv);
    k_deg_rsqrt<<<(NNODES + TB - 1) / TB, TB, 0, stream>>>(dinv);

    // x -> bf16
    {
        long long n = (long long)NNODES * 64;
        k_f2bf<<<(unsigned)((n + TB - 1) / TB), TB, 0, stream>>>(x, actbf, n);
    }

    const int dims[7] = {64, 512, 512, 256, 128, 64, 32};
    const int rowTiles = (NNODES + 15) / 16;           // 3125
    for (int i = 0; i < 6; ++i) {
        int K = dims[i], dout = dims[i + 1];
        int KT = K >> 5, CT = dout >> 4;

        k_repack_w<<<(KT * CT * 32 + TB - 1) / TB, TB, 0, stream>>>(W[i], wpack, K, dout);

        dim3 gg((rowTiles + 7) / 8, (CT + 3) / 4);
        k_wmma_gemm<<<gg, TB, 0, stream>>>(actbf, wpack, hw, K, dout);

        long long nd = (long long)NNODES * dout;
        k_agg_init<<<(unsigned)((nd + TB - 1) / TB), TB, 0, stream>>>(hw, dinv, Bv[i], agg, dout);

        int chunks = dout >> 4;
        int cshift = __builtin_ctz(chunks);
        long long tot = (long long)NEDGES * chunks;
        // grid-stride: ~4 iterations per thread so the next-iteration prefetch has work to hide behind
        long long want = (tot + (long long)TB * 4 - 1) / ((long long)TB * 4);
        unsigned sblocks = (unsigned)(want < 1 ? 1 : want);
        k_scatter<<<sblocks, TB, 0, stream>>>(ei, dinv, hw, agg, dout, cshift);

        k_zero2<<<(dout + TB - 1) / TB, TB, 0, stream>>>(bsum, bsq, dout);
        k_bn_partial<<<1024, TB, 0, stream>>>(agg, bsum, bsq, dout);
        k_bn_final<<<(dout + TB - 1) / TB, TB, 0, stream>>>(bsum, bsq, dout);

        const bf16_t* resid = (i == 1) ? identbf : nullptr;
        bf16_t* obf  = (i < 5) ? actbf : nullptr;
        bf16_t* obf2 = (i == 0) ? identbf : nullptr;
        float*  of32 = (i == 5) ? hw : nullptr;        // last layer: fp32 for the head (hw reusable here)
        k_bn_norm<<<(unsigned)((nd + TB - 1) / TB), TB, 0, stream>>>(
            agg, bsum, bsq, Gv[i], Bt[i], resid, obf, obf2, of32, dout);
    }

    k_global<<<(NNODES + TB - 1) / TB, TB, 0, stream>>>(gf, gW1, gb1, gW2, gb2, g2);
    k_final<<<(NNODES + TB - 1) / TB, TB, 0, stream>>>(hw, g2, fcW, fcb, out);
}
